// _NonLocalBlock_25314537243177
// MI455X (gfx1250) — matrix operand order repaired and checked by reading, not compiled, not run on hardware
//
#include <hip/hip_runtime.h>
#include <hip/hip_bf16.h>

typedef __attribute__((ext_vector_type(16))) __bf16 v16bf;
typedef __attribute__((ext_vector_type(8)))  float  v8f;

#define BB 8
#define CC 64
#define CI 32
#define HH 48
#define WW 48
#define NN (HH*WW)         /* 2304 */
#define NT (NN/16)         /* 144  */
#define KSPLIT 4
#define MCHUNK (NN/KSPLIT) /* 576 K-values per split */
#define INV_N (1.0f/(float)NN)
#define BN_EPS 1e-5f

/* ---- workspace byte offsets ---- */
#define OFF_TVEC  0u
#define OFF_PVEC  256u
#define OFF_CONST 512u
#define OFF_A     1024u
#define OFF_B     (OFF_A  + 4u*BB*NN)
#define OFF_GX    (OFF_B  + 4u*BB*NN)              /* gxT: [b][ci][n] bf16 */
#define OFF_Y     (OFF_GX + 2u*BB*NN*CI)           /* y partials: [ks][b][n][ci] f32 */
#define OFF_WY    (OFF_Y  + 4u*KSPLIT*BB*NN*CI)
#define OFF_STATS (OFF_WY + 4u*BB*CC*NN)

/* Fold theta/phi convs into length-C vectors:
   a[n] = x[:,n] . tvec + tconst,  tvec = th_w^T cp_wt */
__global__ void k_prep(const float* __restrict__ th_w, const float* __restrict__ th_b,
                       const float* __restrict__ ph_w, const float* __restrict__ ph_b,
                       const float* __restrict__ cp_wt, const float* __restrict__ cp_wp,
                       float* __restrict__ tvec, float* __restrict__ pvec,
                       float* __restrict__ consts) {
  int c = threadIdx.x;
  if (c < CC) {
    float t = 0.f, p = 0.f;
    for (int i = 0; i < CI; i++) {
      t += th_w[i*CC + c] * cp_wt[i];
      p += ph_w[i*CC + c] * cp_wp[i];
    }
    tvec[c] = t; pvec[c] = p;
  }
  if (c == 0) {
    float t = 0.f, p = 0.f;
    for (int i = 0; i < CI; i++) { t += th_b[i]*cp_wt[i]; p += ph_b[i]*cp_wp[i]; }
    consts[0] = t; consts[1] = p;
  }
}

/* Per (b,n): g_x row -> TRANSPOSED bf16 buffer gxT[b][ci][n] (WMMA-B friendly),
   plus pre-scaled a[n]/N and bscore[n]/N. */
__global__ __launch_bounds__(256) void k_proj(
    const float* __restrict__ x,
    const float* __restrict__ g_w, const float* __restrict__ g_b,
    const float* __restrict__ tvec, const float* __restrict__ pvec,
    const float* __restrict__ consts,
    __bf16* __restrict__ gxt, float* __restrict__ a_arr, float* __restrict__ b_arr) {
  __shared__ float gw_s[CI*CC];
  __shared__ float gb_s[CI];
  __shared__ float tv_s[CC], pv_s[CC], cs_s[2];
  const int tid = threadIdx.x;
  for (int i = tid; i < CI*CC; i += 256) gw_s[i] = g_w[i];
  if (tid < CI) gb_s[tid] = g_b[tid];
  if (tid < CC) { tv_s[tid] = tvec[tid]; pv_s[tid] = pvec[tid]; }
  if (tid < 2)  cs_s[tid] = consts[tid];
  __syncthreads();

  const int p = blockIdx.x * 256 + tid;       /* p in [0, B*N) */
  const int b = p / NN, n = p % NN;
  float acc[CI];
  #pragma unroll
  for (int i = 0; i < CI; i++) acc[i] = gb_s[i];
  float av = cs_s[0], bv = cs_s[1];
  const float* xb = x + (size_t)b*CC*NN + n;
  for (int c = 0; c < CC; c++) {
    float xv = xb[(size_t)c*NN];              /* coalesced across threads */
    av += xv * tv_s[c];
    bv += xv * pv_s[c];
    #pragma unroll
    for (int i = 0; i < CI; i++) acc[i] += xv * gw_s[i*CC + c];
  }
  __bf16* gxo = gxt + (size_t)b*CI*NN + n;    /* transpose store: stride NN per ci */
  #pragma unroll
  for (int i = 0; i < CI; i++) gxo[(size_t)i*NN] = (__bf16)acc[i];
  a_arr[p] = av * INV_N;                      /* fold 1/N: relu((a+b)/N) */
  b_arr[p] = bv * INV_N;
}

/* Main contraction: y[16 rows x 32 ci] per wave via v_wmma_f32_16x16x32_bf16,
   K split 4 ways across blockIdx.z for occupancy (4608 waves).
   A (16x32 bf16, contiguous k order): lane L -> row M=L&15, half=L>>4;
     elements 0..7  = relu(a_row + bscore[mb + 16*half + e])
     elements 8..15 = relu(a_row + bscore[mb + 16*half + 8 + e])
   B (32x16 bf16): lane L -> col=L&15; elements e = gxT[col][mb + 16*half + e]
     -> one contiguous aligned 32-byte load per operand. */
__global__ __launch_bounds__(32) void k_attn(
    const float* __restrict__ a_arr, const float* __restrict__ b_arr,
    const __bf16* __restrict__ gxt, float* __restrict__ y) {
  const int b    = blockIdx.y;
  const int nb   = blockIdx.x * 16;
  const int ks   = blockIdx.z;
  const int m0   = ks * MCHUNK;
  const int lane = threadIdx.x;
  const int row  = lane & 15;                 /* A row / B-C-D column */
  const int half = lane >> 4;

  const float aval = a_arr[b*NN + nb + row];  /* already scaled by 1/N */
  v8f acc0 = {}; v8f acc1 = {};

  const __bf16* gxt_b = gxt + (size_t)b*CI*NN;
  const __bf16* pB0 = gxt_b + (size_t)row*NN      + 16*half;
  const __bf16* pB1 = gxt_b + (size_t)(row+16)*NN + 16*half;
  const float*  pbs = b_arr + (size_t)b*NN + 16*half;

  #pragma unroll 2
  for (int mb = m0; mb < m0 + MCHUNK; mb += 32) {
    v8f blo = *(const v8f*)(pbs + mb);        /* bscore[mb+16h .. +7]     */
    v8f bhi = *(const v8f*)(pbs + mb + 8);    /* bscore[mb+16h+8 .. +7]   */
    v16bf B0 = *(const v16bf*)(pB0 + mb);     /* 32B aligned contiguous   */
    v16bf B1 = *(const v16bf*)(pB1 + mb);
    if (mb + 32 < m0 + MCHUNK)
      __builtin_prefetch(pB0 + mb + 32, 0, 3);

    v16bf A;
    #pragma unroll
    for (int e = 0; e < 8; e++) {
      A[e]     = (__bf16)fmaxf(aval + blo[e], 0.0f);
      A[e + 8] = (__bf16)fmaxf(aval + bhi[e], 0.0f);
    }
    acc0 = __builtin_amdgcn_wmma_f32_16x16x32_bf16(false, A, false, B0, (short)0, acc0, false, false);
    acc1 = __builtin_amdgcn_wmma_f32_16x16x32_bf16(false, A, false, B1, (short)0, acc1, false, false);
  }

  /* C/D layout: VGPR v -> row v + 8*half, col = lane&15 */
  float* yo = y + ((size_t)ks*BB*NN + (size_t)b*NN + nb) * CI;
  #pragma unroll
  for (int v = 0; v < 8; v++) {
    const int rw = v + 8*half;
    yo[(size_t)rw*CI + row]      = acc0[v];
    yo[(size_t)rw*CI + row + 16] = acc1[v];
  }
}

/* wy[b,c,n] = W_w[c,:] . (sum of K-split y partials)[b,n,:] + W_b[c] */
__global__ __launch_bounds__(256) void k_wy(
    const float* __restrict__ y, const float* __restrict__ W_w,
    const float* __restrict__ W_b, float* __restrict__ wy) {
  __shared__ float ww_s[CC*CI];
  __shared__ float wb_s[CC];
  const int tid = threadIdx.x;
  for (int i = tid; i < CC*CI; i += 256) ww_s[i] = W_w[i];
  if (tid < CC) wb_s[tid] = W_b[tid];
  __syncthreads();

  const int p = blockIdx.x * 256 + tid;
  const int b = p / NN, n = p % NN;
  float yv[CI];
  const float* y0 = y + (size_t)p * CI;
  #pragma unroll
  for (int i = 0; i < CI; i++) {
    float s = y0[i];
    #pragma unroll
    for (int k = 1; k < KSPLIT; k++) s += y0[(size_t)k*BB*NN*CI + i];
    yv[i] = s;
  }
  float* wyb = wy + (size_t)b*CC*NN + n;
  for (int c = 0; c < CC; c++) {
    float s = wb_s[c];
    #pragma unroll
    for (int i = 0; i < CI; i++) s += ww_s[c*CI + i] * yv[i];
    wyb[(size_t)c*NN] = s;
  }
}

/* Training-mode BN batch stats per channel (biased var, matching jnp.var). */
__global__ __launch_bounds__(256) void k_stats(const float* __restrict__ wy,
                                               float* __restrict__ stats) {
  const int c = blockIdx.x;
  const int tid = threadIdx.x;
  float s = 0.f, s2 = 0.f;
  for (int j = tid; j < BB*NN; j += 256) {
    const int b = j / NN, n = j % NN;
    float v = wy[((size_t)b*CC + c)*NN + n];
    s += v; s2 += v*v;
  }
  __shared__ float rs[256], rs2[256];
  rs[tid] = s; rs2[tid] = s2;
  __syncthreads();
  for (int off = 128; off > 0; off >>= 1) {
    if (tid < off) { rs[tid] += rs[tid + off]; rs2[tid] += rs2[tid + off]; }
    __syncthreads();
  }
  if (tid == 0) {
    const float cnt = (float)(BB*NN);
    float mean = rs[0] / cnt;
    float var  = rs2[0] / cnt - mean*mean;
    stats[c] = mean; stats[64 + c] = var;
  }
}

/* out = (wy - mean) * rsqrt(var+eps) * bn_g + bn_b + x */
__global__ __launch_bounds__(256) void k_final(
    const float* __restrict__ wy, const float* __restrict__ x,
    const float* __restrict__ stats,
    const float* __restrict__ bn_g, const float* __restrict__ bn_b,
    float* __restrict__ out) {
  const size_t idx = (size_t)blockIdx.x * 256 + threadIdx.x;
  const int c = (int)((idx / NN) % CC);
  const float mean = stats[c], var = stats[64 + c];
  float v = (wy[idx] - mean) * rsqrtf(var + BN_EPS);
  out[idx] = v * bn_g[c] + bn_b[c] + x[idx];
}

extern "C" void kernel_launch(void* const* d_in, const int* in_sizes, int n_in,
                              void* d_out, int out_size, void* d_ws, size_t ws_size,
                              hipStream_t stream) {
  const float* x     = (const float*)d_in[0];
  const float* g_w   = (const float*)d_in[1];
  const float* g_b   = (const float*)d_in[2];
  const float* th_w  = (const float*)d_in[3];
  const float* th_b  = (const float*)d_in[4];
  const float* ph_w  = (const float*)d_in[5];
  const float* ph_b  = (const float*)d_in[6];
  const float* cp_wt = (const float*)d_in[7];
  const float* cp_wp = (const float*)d_in[8];
  const float* W_w   = (const float*)d_in[9];
  const float* W_b   = (const float*)d_in[10];
  const float* bn_g  = (const float*)d_in[11];
  const float* bn_b  = (const float*)d_in[12];
  float* out = (float*)d_out;

  char* ws = (char*)d_ws;
  float*  tvec  = (float*)(ws + OFF_TVEC);
  float*  pvec  = (float*)(ws + OFF_PVEC);
  float*  cst   = (float*)(ws + OFF_CONST);
  float*  a_arr = (float*)(ws + OFF_A);
  float*  b_arr = (float*)(ws + OFF_B);
  __bf16* gxt   = (__bf16*)(ws + OFF_GX);
  float*  y     = (float*)(ws + OFF_Y);
  float*  wy    = (float*)(ws + OFF_WY);
  float*  stats = (float*)(ws + OFF_STATS);

  k_prep<<<1, 64, 0, stream>>>(th_w, th_b, ph_w, ph_b, cp_wt, cp_wp, tvec, pvec, cst);
  k_proj<<<(BB*NN)/256, 256, 0, stream>>>(x, g_w, g_b, tvec, pvec, cst, gxt, a_arr, b_arr);
  k_attn<<<dim3(NT, BB, KSPLIT), 32, 0, stream>>>(a_arr, b_arr, gxt, y);
  k_wy<<<(BB*NN)/256, 256, 0, stream>>>(y, W_w, W_b, wy);
  k_stats<<<CC, 256, 0, stream>>>(wy, stats);
  k_final<<<(BB*CC*NN)/256, 256, 0, stream>>>(wy, x, stats, bn_g, bn_b, out);
}
